// LoRAExpert_88630945120687
// MI455X (gfx1250) — compile-verified, matrix operand order checked by reading, not run on hardware
//
#include <hip/hip_runtime.h>

// Problem constants (from reference)
#define T_TOK 16384
#define NE    8
#define DIN   1024
#define DOUT  1024
#define NA    8
#define RR    16
#define KLORA 128              // NA * RR : folded LoRA K-extension
#define KTOT  (DIN + KLORA)    // 1152

// CDNA5 async global->LDS path (ASYNCcnt). Fallback = sync VGPR round trip.
#if defined(__AMDGCN__) && __has_builtin(__builtin_amdgcn_global_load_async_to_lds_b128)
#define ASYNC_LDS 1
#else
#define ASYNC_LDS 0
#endif

typedef __attribute__((ext_vector_type(16))) __bf16       v16bf;
typedef __attribute__((ext_vector_type(8)))  float        v8f;
typedef __attribute__((ext_vector_type(4)))  unsigned int u32x4;
typedef __attribute__((ext_vector_type(4)))  int          i32x4;

struct Frag32B { u32x4 lo; u32x4 hi; };   // 32 bytes == v16bf

__device__ __forceinline__ unsigned short f2bf(float f) {
  unsigned int u = __builtin_bit_cast(unsigned int, f);
  u += 0x7FFFu + ((u >> 16) & 1u);
  return (unsigned short)(u >> 16);
}

__device__ __forceinline__ v16bf make_frag(const unsigned short* p, int o0, int o1) {
  Frag32B f;
  f.lo = *(const u32x4*)(p + o0);   // ds_load_b128
  f.hi = *(const u32x4*)(p + o1);   // ds_load_b128
  return __builtin_bit_cast(v16bf, f);
}

// Stage a 128-row x 32-half (64B) tile from global (row stride `rstride` halves)
// into an LDS region with stride 40 halves (80B, keeps 16B alignment).
// Rows >= maxr are clamped (duplicate data; consumers mask those outputs).
__device__ __forceinline__ void stage_tile(const unsigned short* __restrict__ g,
                                           int rstride, int maxr,
                                           unsigned short* l, int tid) {
  #pragma unroll
  for (int s = tid; s < 512; s += 256) {
    int r  = s >> 2;
    int sk = (s & 3) * 8;
    int rr = r < maxr ? r : (maxr - 1);
    const unsigned short* gp = g + (long long)rr * rstride + sk;
    unsigned short* lp = l + r * 40 + sk;
#if ASYNC_LDS
    // per-lane 16B global -> LDS, tracked by ASYNCcnt (no VGPR data round trip)
    __builtin_amdgcn_global_load_async_to_lds_b128(
        (__attribute__((address_space(1))) i32x4*)(unsigned long long)gp,
        (__attribute__((address_space(3))) i32x4*)(unsigned int)(unsigned long long)lp,
        0, 0);
#else
    *(u32x4*)lp = *(const u32x4*)gp;
#endif
  }
}

__device__ __forceinline__ void wait_stage() {
#if ASYNC_LDS
  asm volatile("s_wait_asynccnt 0x0" ::: "memory");
#endif
  __syncthreads();
}

// ---------------------------------------------------------------------------
// Kernel 1: routing setup. wsI[0..8]=expert prefix offsets, wsI[16]=ntiles,
// wsI[32..]={row_start,row_count,expert,0}*ntiles  (ntiles <= 136)
// ---------------------------------------------------------------------------
__global__ void k_setup(const int* __restrict__ gs, int* __restrict__ wsI) {
  if (threadIdx.x != 0 || blockIdx.x != 0) return;
  int off = 0;
  wsI[0] = 0;
  for (int e = 0; e < NE; ++e) { off += gs[e]; wsI[1 + e] = off; }
  int* tiles = wsI + 32;
  int cnt = 0, start = 0;
  for (int e = 0; e < NE; ++e) {
    int end = wsI[1 + e];
    for (int s = start; s < end; s += 128) {
      int rc = end - s; if (rc > 128) rc = 128;
      tiles[cnt * 4 + 0] = s;
      tiles[cnt * 4 + 1] = rc;
      tiles[cnt * 4 + 2] = e;
      tiles[cnt * 4 + 3] = 0;
      ++cnt;
    }
    start = end;
  }
  wsI[16] = cnt;
}

// ---------------------------------------------------------------------------
// Kernel 2: fp32->bf16 conversions + layout transforms (one-time pass):
//   xb    [T][DIN]              = bf16(x)
//   wbT   [E][DOUT n][DIN k]    = bf16(weight[e][k][n])     (N-major for B)
//   bcatT [E][DOUT n][KLORA s]  = bf16(lora_B[a][e][r][n]), s = a*16+r
//   acatT [E][KLORA s][DIN k]   = bf16(lora_A[a][e][k][r]), s = a*16+r
// ---------------------------------------------------------------------------
__global__ __launch_bounds__(256) void k_convert(
    const float* __restrict__ x, const float* __restrict__ w,
    const float* __restrict__ lA, const float* __restrict__ lB,
    unsigned short* __restrict__ xb, unsigned short* __restrict__ wbT,
    unsigned short* __restrict__ bcatT, unsigned short* __restrict__ acatT) {
  const int stride = gridDim.x * blockDim.x;
  const int gid = blockIdx.x * blockDim.x + threadIdx.x;
  for (int i = gid; i < T_TOK * DIN; i += stride) xb[i] = f2bf(x[i]);
  for (int i = gid; i < NE * DOUT * DIN; i += stride) {
    int e = i >> 20, rem = i & 1048575, n = rem >> 10, k = i & 1023;
    wbT[i] = f2bf(w[((long long)e << 20) + k * 1024 + n]);
  }
  for (int i = gid; i < NE * DOUT * KLORA; i += stride) {
    int e = i >> 17, rem = i & 131071, n = rem >> 7, s = i & 127;
    int a = s >> 4, r = s & 15;
    bcatT[i] = f2bf(lB[(((long long)(a * NE + e) * RR) + r) * DOUT + n]);
  }
  for (int i = gid; i < NE * KLORA * DIN; i += stride) {
    int e = i >> 17, rem = i & 131071, s = rem >> 10, k = i & 1023;
    int a = s >> 4, r = s & 15;
    acatT[i] = f2bf(lA[(((long long)(a * NE + e) * DIN) + k) * RR + r]);
  }
}

// ---------------------------------------------------------------------------
// Kernel 3: LoRA intermediate as a WMMA grouped GEMM:
//   inter_full[t, s] = (x_t . lora_A[a',e_t,:,r]) for ALL a' (s = a'*16+r),
// epilogue masks to the token's adapter, scales, emits bf16 interA[T][128].
// Tile: 128 rows x 128 cols, K = 1024, double-buffered async staging.
// ---------------------------------------------------------------------------
__global__ __launch_bounds__(256) void k_lora_inter(
    const unsigned short* __restrict__ xb,
    const unsigned short* __restrict__ acatT,
    const int* __restrict__ adap, const float* __restrict__ scal,
    const int* __restrict__ wsI, unsigned short* __restrict__ interA) {
  __shared__ __align__(16) unsigned short lsA[2][128 * 40];
  __shared__ __align__(16) unsigned short lsB[2][128 * 40];

  const int ntiles = wsI[16];
  const int mt = blockIdx.x;
  if (mt >= ntiles) return;
  const int* tile = wsI + 32 + mt * 4;
  const int row0 = tile[0];
  const int rcnt = tile[1];
  const int e    = tile[2];

  const int tid  = threadIdx.x;
  const int lane = tid & 31;
  const int wm   = (tid >> 5) >> 2;
  const int wn   = (tid >> 5) & 3;
  const int lhi  = lane >> 4;
  const int l15  = lane & 15;
  const int maxr = T_TOK - row0;

  const unsigned short* aBase = xb + (long long)row0 * DIN;
  const unsigned short* bBase = acatT + (long long)e * KLORA * DIN;

  stage_tile(aBase, DIN, maxr, lsA[0], tid);
  stage_tile(bBase, DIN, 128,  lsB[0], tid);

  v8f acc[4][2] = {};
  int buf = 0;
  for (int kt = 0; kt < DIN; kt += 32) {
    wait_stage();
    if (kt + 32 < DIN) {
      stage_tile(aBase + kt + 32, DIN, maxr, lsA[buf ^ 1], tid);
      stage_tile(bBase + kt + 32, DIN, 128,  lsB[buf ^ 1], tid);
    }
    const int kq = lhi * 8;
    v16bf af[4], bfr[2];
    #pragma unroll
    for (int mi = 0; mi < 4; ++mi) {
      int r = wm * 64 + mi * 16 + l15;
      af[mi] = make_frag(lsA[buf], r * 40 + kq, r * 40 + 16 + kq);
    }
    #pragma unroll
    for (int ni = 0; ni < 2; ++ni) {
      int c = wn * 32 + ni * 16 + l15;
      bfr[ni] = make_frag(lsB[buf], c * 40 + kq, c * 40 + 16 + kq);
    }
    #pragma unroll
    for (int mi = 0; mi < 4; ++mi)
      #pragma unroll
      for (int ni = 0; ni < 2; ++ni)
        acc[mi][ni] = __builtin_amdgcn_wmma_f32_16x16x32_bf16(
            false, af[mi], false, bfr[ni], (short)0, acc[mi][ni], false, false);
    buf ^= 1;
  }

  // mask to token's adapter, scale, store bf16
  #pragma unroll
  for (int mi = 0; mi < 4; ++mi) {
    #pragma unroll
    for (int j = 0; j < 8; ++j) {
      int ml = wm * 64 + mi * 16 + lhi * 8 + j;
      if (ml < rcnt) {
        int t = row0 + ml;
        int a = adap[t];
        float sc = scal[a];
        #pragma unroll
        for (int ni = 0; ni < 2; ++ni) {
          int s = wn * 32 + ni * 16 + l15;
          float v = ((s >> 4) == a) ? sc * acc[mi][ni][j] : 0.f;
          interA[(long long)t * KLORA + s] = f2bf(v);
        }
      }
    }
  }
}

// ---------------------------------------------------------------------------
// Kernel 4: main grouped GEMM, K = 1024 (base) + 128 (folded LoRA).
// 128x128 tile / block, 8 wave32s (2Mx4N), 4x2 WMMA subtiles per wave,
// double-buffered async LDS staging; plain f32 store epilogue.
// ---------------------------------------------------------------------------
__global__ __launch_bounds__(256) void k_gemm(
    const unsigned short* __restrict__ xb,
    const unsigned short* __restrict__ interA,
    const unsigned short* __restrict__ wbT,
    const unsigned short* __restrict__ bcatT,
    const int* __restrict__ wsI,
    float* __restrict__ out) {
  __shared__ __align__(16) unsigned short lsA[2][128 * 40];
  __shared__ __align__(16) unsigned short lsB[2][128 * 40];

  const int ntiles = wsI[16];
  const int mt = blockIdx.x;
  if (mt >= ntiles) return;
  const int* tile = wsI + 32 + mt * 4;
  const int row0 = tile[0];
  const int rcnt = tile[1];
  const int e    = tile[2];
  const int n0   = blockIdx.y * 128;

  const int tid  = threadIdx.x;
  const int lane = tid & 31;
  const int wm   = (tid >> 5) >> 2;
  const int wn   = (tid >> 5) & 3;
  const int lhi  = lane >> 4;
  const int l15  = lane & 15;
  const int maxr = T_TOK - row0;

  // K-phase dependent tile sources (base vs folded-LoRA extension)
  const unsigned short* aB0 = xb     + (long long)row0 * DIN;
  const unsigned short* aB1 = interA + (long long)row0 * KLORA;
  const unsigned short* bB0 = wbT   + (long long)e * DOUT * DIN   + (long long)n0 * DIN;
  const unsigned short* bB1 = bcatT + (long long)e * DOUT * KLORA + (long long)n0 * KLORA;

  stage_tile(aB0, DIN, maxr, lsA[0], tid);
  stage_tile(bB0, DIN, 128,  lsB[0], tid);

  v8f acc[4][2] = {};
  int buf = 0;
  for (int kt = 0; kt < KTOT; kt += 32) {
    wait_stage();
    const int nkt = kt + 32;
    if (nkt < KTOT) {
      if (nkt < DIN) {
        stage_tile(aB0 + nkt, DIN, maxr, lsA[buf ^ 1], tid);
        stage_tile(bB0 + nkt, DIN, 128,  lsB[buf ^ 1], tid);
      } else {
        stage_tile(aB1 + (nkt - DIN), KLORA, maxr, lsA[buf ^ 1], tid);
        stage_tile(bB1 + (nkt - DIN), KLORA, 128,  lsB[buf ^ 1], tid);
      }
    }
    const int kq = lhi * 8;
    v16bf af[4], bfr[2];
    #pragma unroll
    for (int mi = 0; mi < 4; ++mi) {
      int r = wm * 64 + mi * 16 + l15;
      af[mi] = make_frag(lsA[buf], r * 40 + kq, r * 40 + 16 + kq);
    }
    #pragma unroll
    for (int ni = 0; ni < 2; ++ni) {
      int c = wn * 32 + ni * 16 + l15;
      bfr[ni] = make_frag(lsB[buf], c * 40 + kq, c * 40 + 16 + kq);
    }
    #pragma unroll
    for (int mi = 0; mi < 4; ++mi)
      #pragma unroll
      for (int ni = 0; ni < 2; ++ni)
        acc[mi][ni] = __builtin_amdgcn_wmma_f32_16x16x32_bf16(
            false, af[mi], false, bfr[ni], (short)0, acc[mi][ni], false, false);
    buf ^= 1;
  }

  #pragma unroll
  for (int mi = 0; mi < 4; ++mi) {
    #pragma unroll
    for (int ni = 0; ni < 2; ++ni) {
      int col = n0 + wn * 32 + ni * 16 + l15;
      #pragma unroll
      for (int j = 0; j < 8; ++j) {
        int ml = wm * 64 + mi * 16 + lhi * 8 + j;
        if (ml < rcnt)
          out[(long long)(row0 + ml) * DOUT + col] = acc[mi][ni][j];
      }
    }
  }
}

// ---------------------------------------------------------------------------
// Host-side launch. Inputs (reference order):
// 0: x [T,IN] f32  1: group_sizes [E] i32  2: adapter_indices_sorted [T] i32
// 3: weight [E,IN,OUT] f32  4: lora_A [A,E,IN,R] f32
// 5: lora_B [A,E,R,OUT] f32 6: lora_scaling [A] f32
// ---------------------------------------------------------------------------
extern "C" void kernel_launch(void* const* d_in, const int* in_sizes, int n_in,
                              void* d_out, int out_size, void* d_ws, size_t ws_size,
                              hipStream_t stream) {
  (void)in_sizes; (void)n_in; (void)out_size; (void)ws_size;
  const float* x    = (const float*)d_in[0];
  const int*   gs   = (const int*)d_in[1];
  const int*   adap = (const int*)d_in[2];
  const float* w    = (const float*)d_in[3];
  const float* lA   = (const float*)d_in[4];
  const float* lB   = (const float*)d_in[5];
  const float* scal = (const float*)d_in[6];
  float*       out  = (float*)d_out;

  char* ws = (char*)d_ws;
  int* wsI = (int*)ws;
  size_t off = 4096;
  unsigned short* xb     = (unsigned short*)(ws + off); off += (size_t)T_TOK * DIN * 2;
  unsigned short* wbT    = (unsigned short*)(ws + off); off += (size_t)NE * DOUT * DIN * 2;
  unsigned short* bcatT  = (unsigned short*)(ws + off); off += (size_t)NE * DOUT * KLORA * 2;
  unsigned short* acatT  = (unsigned short*)(ws + off); off += (size_t)NE * KLORA * DIN * 2;
  unsigned short* interA = (unsigned short*)(ws + off);

  k_setup<<<1, 32, 0, stream>>>(gs, wsI);
  k_convert<<<2048, 256, 0, stream>>>(x, w, lA, lB, xb, wbT, bcatT, acatT);
  k_lora_inter<<<136, 256, 0, stream>>>(xb, acatT, adap, scal, wsI, interA);
  k_gemm<<<dim3(136, DOUT / 128), 256, 0, stream>>>(xb, interA, wbT, bcatT, wsI, out);
}
